// Linear_55576876810249
// MI455X (gfx1250) — matrix operand order repaired and checked by reading, not compiled, not run on hardware
//
#include <hip/hip_runtime.h>

// 3x3 same-conv as im2col-GEMM on gfx1250 WMMA.
// M = 64 out-channels, N = pixels, K = 288 (reordered k' = tap*32 + c).
// f32 -> f16 staged in LDS, f32 accumulation via v_wmma_f32_16x16x32_f16.
// Roofline: 38.7 GFLOP / ~0.4 GB HBM -> memory-bound at ~17us with f16 WMMA;
// f32 WMMA (16x16x4) would be ~8x compute-bound, so f16 is the right path.

typedef __attribute__((ext_vector_type(16))) _Float16     v16h;
typedef __attribute__((ext_vector_type(8)))  float        v8f;
typedef __attribute__((ext_vector_type(4)))  unsigned int v4u;

#define C_IN   32
#define C_OUT  64
#define H_IMG  256
#define W_IMG  256
#define TH     4            // output rows per workgroup
#define TW     32           // output width per workgroup
#define XS_ROWS (TH + 2)    // 6 rows incl. halo
#define XS_W    (TW + 2)    // 34 valid width incl. halo
#define XS_WPAD 37          // padded dword pitch (bank-conflict avoidance)
#define WS_ROW  148         // padded dwords per out-channel weight row (144 used)

#define XS_DWORDS (16 * XS_ROWS * XS_WPAD)  // 16 channel-pairs -> 3552
#define WS_DWORDS (C_OUT * WS_ROW)          // 9472
// total LDS = (3552 + 9472)*4 = 52096 B per workgroup

__global__ __launch_bounds__(256)
void conv3x3_wmma_f16(const float* __restrict__ x,
                      const float* __restrict__ weight,
                      const float* __restrict__ bias,
                      float* __restrict__ out)
{
    __shared__ unsigned int xs[XS_DWORDS];  // x tile: f16 pairs (c even lo, c odd hi)
    __shared__ unsigned int ws[WS_DWORDS];  // weight: f16 pairs along k' = tap*32 + c

    const int tid  = threadIdx.x;
    const int lane = tid & 31;
    const int wave = tid >> 5;
    const int g    = lane >> 4;   // lane half-group (K-split select)
    const int p    = lane & 15;   // matrix row/col select within 16

    const int tilesPerImg = (H_IMG / TH) * (W_IMG / TW);     // 64*8 = 512
    const int b  = blockIdx.x / tilesPerImg;
    const int t  = blockIdx.x % tilesPerImg;
    const int h0 = (t / (W_IMG / TW)) * TH;
    const int w0 = (t % (W_IMG / TW)) * TW;

    // ---- stage weight into LDS: reorder (o, c*9+tap) -> (o, tap*32+c), f32->f16 pairs
    for (int q = tid; q < C_OUT * 144; q += 256) {
        int o   = q / 144;
        int qq  = q - o * 144;
        int tap = qq >> 4;          // 0..8 (kh*3+kw)
        int cp  = qq & 15;          // channel pair index
        int c0  = cp * 2;
        float f0 = weight[o * 288 + c0 * 9 + tap];
        float f1 = weight[o * 288 + (c0 + 1) * 9 + tap];
        _Float16 h0 = (_Float16)f0;
        _Float16 h1 = (_Float16)f1;
        unsigned u = (unsigned)__builtin_bit_cast(unsigned short, h0)
                   | ((unsigned)__builtin_bit_cast(unsigned short, h1) << 16);
        ws[o * WS_ROW + qq] = u;
    }

    // ---- stage x tile (6 rows incl. halo, zero-padded, f32->f16) ----
    _Float16* xsh = (_Float16*)xs;
    for (int idx = tid; idx < C_IN * XS_ROWS * XS_W; idx += 256) {
        int c   = idx / (XS_ROWS * XS_W);
        int rem = idx - c * (XS_ROWS * XS_W);
        int r   = rem / XS_W;
        int wl  = rem - r * XS_W;
        int gr  = h0 - 1 + r;
        int gw  = w0 - 1 + wl;
        float v = 0.0f;
        if ((unsigned)gr < H_IMG && (unsigned)gw < W_IMG)
            v = x[(((size_t)b * C_IN + c) * H_IMG + gr) * W_IMG + gw];
        xsh[(((c >> 1) * XS_ROWS + r) * XS_WPAD + wl) * 2 + (c & 1)] = (_Float16)v;
    }

    __syncthreads();

    // wave -> one 16-pixel N-subtile (row hh, half wseg), all 4 M-subtiles
    const int hh   = wave >> 1;        // 0..3
    const int wseg = wave & 1;         // 0..1
    const int ww   = wseg * 16 + p;    // pixel column within tile (0..31)

    v8f acc[4] = {v8f{}, v8f{}, v8f{}, v8f{}};

    union AF { v4u u[2]; v16h h; };
    union BF { unsigned int u[8]; v16h h; };

    #pragma unroll
    for (int j = 0; j < 9; ++j) {      // K-chunk = tap j, 32 channels
        const int kh = j / 3, kw = j % 3;
        const int r  = hh + kh;        // 0..5
        const int w  = ww + kw;        // 0..33

        // B fragment (32x16 K x N): lane group g holds K = g*16 + i; dword d = channels g*8+d pair
        BF bf;
        #pragma unroll
        for (int d = 0; d < 8; ++d)
            bf.u[d] = xs[((g * 8 + d) * XS_ROWS + r) * XS_WPAD + w];

        #pragma unroll
        for (int m = 0; m < 4; ++m) {
            const int o = m * 16 + p;  // A: lane holds M = lane%16
            AF af;
            af.u[0] = *(const v4u*)&ws[o * WS_ROW + j * 16 + g * 8];      // K = g*16 + 0..7
            af.u[1] = *(const v4u*)&ws[o * WS_ROW + j * 16 + g * 8 + 4];  // K = g*16 + 8..15
            acc[m] = __builtin_amdgcn_wmma_f32_16x16x32_f16(
                         false, af.h, false, bf.h, (short)0, acc[m], false, false);
        }
    }

    // ---- store: D vgpr v <-> M = v + 8*g, N = lane%16 ----
    const int oh = h0 + hh;
    const int ow = w0 + ww;
    #pragma unroll
    for (int m = 0; m < 4; ++m) {
        #pragma unroll
        for (int v = 0; v < 8; ++v) {
            int o = m * 16 + g * 8 + v;
            float val = acc[m][v] + bias[o];
            out[(((size_t)b * C_OUT + o) * H_IMG + oh) * W_IMG + ow] = val;
        }
    }
}

extern "C" void kernel_launch(void* const* d_in, const int* in_sizes, int n_in,
                              void* d_out, int out_size, void* d_ws, size_t ws_size,
                              hipStream_t stream) {
    const float* x      = (const float*)d_in[0];
    const float* weight = (const float*)d_in[1];
    const float* bias   = (const float*)d_in[2];
    float* out          = (float*)d_out;

    const int B = in_sizes[0] / (C_IN * H_IMG * W_IMG);   // 16
    const int tilesPerImg = (H_IMG / TH) * (W_IMG / TW);  // 512
    dim3 grid(B * tilesPerImg);                           // 8192 workgroups
    dim3 block(256);                                      // 8 waves (wave32)
    conv3x3_wmma_f16<<<grid, block, 0, stream>>>(x, weight, bias, out);
}
